// torchLSTM_cell_my_implementation_48258252538450
// MI455X (gfx1250) — compile-verified, matrix operand order checked by reading, not run on hardware
//
#include <hip/hip_runtime.h>
#include <hip/hip_bf16.h>
#include <math.h>

// ---------------------------------------------------------------------------
// Problem constants (reference: NT=365, NGRID=512, NX=16, NY=1, H=512)
// ---------------------------------------------------------------------------
#define NT   365
#define BSZ  512
#define HID  512
#define NX   16
#define G4H  2048               // 4*H
#define PLANE (BSZ * HID)       // 262144 elements per [B,H] plane

typedef __attribute__((ext_vector_type(16))) __bf16 v16bf;
typedef __attribute__((ext_vector_type(8)))  __bf16 v8bf;
typedef __attribute__((ext_vector_type(8)))  float  v8f;

// ---------------------------------------------------------------------------
// Fragment loaders.
// CDNA5 16-bit A/B fragment (16x32 / 32x16): lane L = (r = L&15, half = L>>4).
//   elements 0..7  hold K = k0 + half*8 + j
//   elements 8..15 hold K = k0 + 16 + half*8 + j
// For C = A * B^T with row-major A[M,K] and row-major Brm[N,K], BOTH operands
// load per-lane-contiguous 16B chunks along K — no transpose needed.
// ---------------------------------------------------------------------------
__device__ __forceinline__ v16bf frag_ld_bf16(const __bf16* p) {
  union { v16bf v; v8bf h[2]; } u;
  u.h[0] = *(const v8bf*)(p);
  u.h[1] = *(const v8bf*)(p + 16);
  return u.v;
}

__device__ __forceinline__ v16bf frag_ld_f32(const float* p) {
  // same K mapping, source is fp32 (G1 gate buffer), convert to bf16 inline
  v16bf f;
  const float4* q0 = (const float4*)(p);
  const float4* q1 = (const float4*)(p + 16);
  float4 a = q0[0], b = q0[1], cc = q1[0], d = q1[1];
  f[0]=(__bf16)a.x; f[1]=(__bf16)a.y; f[2]=(__bf16)a.z; f[3]=(__bf16)a.w;
  f[4]=(__bf16)b.x; f[5]=(__bf16)b.y; f[6]=(__bf16)b.z; f[7]=(__bf16)b.w;
  f[8]=(__bf16)cc.x; f[9]=(__bf16)cc.y; f[10]=(__bf16)cc.z; f[11]=(__bf16)cc.w;
  f[12]=(__bf16)d.x; f[13]=(__bf16)d.y; f[14]=(__bf16)d.z; f[15]=(__bf16)d.w;
  return f;
}

__device__ __forceinline__ float sigm(float x) { return 1.0f / (1.0f + expf(-x)); }

// ---------------------------------------------------------------------------
// Fused gate GEMM:  C[m,n] = bias[n] + sum_k Acat[m,k] * Brm[n,k]
// Acat rows are the concatenation [A0 row | A1 row] (K = KA0 + KA0), which
// implements  x_t @ W_ih^T + h @ W_hh^T  with pre-packed Brm = [W_ih|W_hh].
// Block: 256 thr = 8 waves, block tile 128x128, wave tile 32x64 (2x4 WMMA).
// ---------------------------------------------------------------------------
__global__ __launch_bounds__(256)
void gemm_fused_bt_kernel(const __bf16* __restrict__ A0,
                          const __bf16* __restrict__ A1,
                          int KA0, int lda,
                          const __bf16* __restrict__ Bm, int ldb,
                          const float* __restrict__ bias,
                          float* __restrict__ C, int ldc, int K)
{
  const int lane = threadIdx.x & 31;
  const int wave = threadIdx.x >> 5;
  const int r    = lane & 15;
  const int half = lane >> 4;
  const int m0 = blockIdx.y * 128 + (wave >> 1) * 32;
  const int n0 = blockIdx.x * 128 + (wave & 1) * 64;

  v8f acc[2][4];
#pragma unroll
  for (int ni = 0; ni < 4; ++ni) {
    float bv = bias ? bias[n0 + ni * 16 + r] : 0.0f;
#pragma unroll
    for (int mi = 0; mi < 2; ++mi)
#pragma unroll
      for (int g = 0; g < 8; ++g) acc[mi][ni][g] = bv;
  }

  for (int k0 = 0; k0 < K; k0 += 32) {
    const __bf16* Ab = (k0 < KA0) ? A0 : A1;
    const int kk = (k0 < KA0) ? k0 : (k0 - KA0);

    v16bf af[2];
#pragma unroll
    for (int mi = 0; mi < 2; ++mi)
      af[mi] = frag_ld_bf16(Ab + (size_t)(m0 + mi * 16 + r) * lda + kk + half * 8);

    v16bf bfv[4];
#pragma unroll
    for (int ni = 0; ni < 4; ++ni)
      bfv[ni] = frag_ld_bf16(Bm + (size_t)(n0 + ni * 16 + r) * ldb + k0 + half * 8);

#pragma unroll
    for (int mi = 0; mi < 2; ++mi)
#pragma unroll
      for (int ni = 0; ni < 4; ++ni)
        acc[mi][ni] = __builtin_amdgcn_wmma_f32_16x16x32_bf16(
            false, af[mi], false, bfv[ni], (short)0, acc[mi][ni], false, false);
  }

  // C/D layout: VGPR g, lane (r,half) -> element (M = half*8+g, N = r)
#pragma unroll
  for (int mi = 0; mi < 2; ++mi)
#pragma unroll
    for (int ni = 0; ni < 4; ++ni)
#pragma unroll
      for (int g = 0; g < 8; ++g)
        C[(size_t)(m0 + mi * 16 + half * 8 + g) * ldc + n0 + ni * 16 + r] =
            acc[mi][ni][g];
}

// ---------------------------------------------------------------------------
// Gate-mixing GEMM (z = 0..3 over i/f/c/o):
//   S_z[m,n] = sum_k gate0_z[m,k] * G1[n, z*512 + k]   (= i0 @ ir^T etc.)
// Writes activated values (sigmoid / tanh for z==2) and per-block partial
// sum / sum^2 of the PRE-activation values (for jnp.var of si/sf/sc/so).
// ---------------------------------------------------------------------------
__global__ __launch_bounds__(256)
void sgemm_mix_kernel(const __bf16* __restrict__ g0,
                      const float* __restrict__ G1,
                      float* __restrict__ Sact,
                      float* __restrict__ psum,
                      float* __restrict__ psumsq)
{
  const int z = blockIdx.z;
  const __bf16* A  = g0 + (size_t)z * PLANE;           // [512,512] bf16, ld 512
  const float*  Bm = G1 + z * HID;                      // rows of len 512, ld 2048
  const int lane = threadIdx.x & 31;
  const int wave = threadIdx.x >> 5;
  const int r    = lane & 15;
  const int half = lane >> 4;
  const int m0 = blockIdx.y * 128 + (wave >> 1) * 32;
  const int n0 = blockIdx.x * 128 + (wave & 1) * 64;

  v8f acc[2][4];
#pragma unroll
  for (int mi = 0; mi < 2; ++mi)
#pragma unroll
    for (int ni = 0; ni < 4; ++ni)
#pragma unroll
      for (int g = 0; g < 8; ++g) acc[mi][ni][g] = 0.0f;

  for (int k0 = 0; k0 < HID; k0 += 32) {
    v16bf af[2];
#pragma unroll
    for (int mi = 0; mi < 2; ++mi)
      af[mi] = frag_ld_bf16(A + (size_t)(m0 + mi * 16 + r) * HID + k0 + half * 8);
    v16bf bfv[4];
#pragma unroll
    for (int ni = 0; ni < 4; ++ni)
      bfv[ni] = frag_ld_f32(Bm + (size_t)(n0 + ni * 16 + r) * G4H + k0 + half * 8);
#pragma unroll
    for (int mi = 0; mi < 2; ++mi)
#pragma unroll
      for (int ni = 0; ni < 4; ++ni)
        acc[mi][ni] = __builtin_amdgcn_wmma_f32_16x16x32_bf16(
            false, af[mi], false, bfv[ni], (short)0, acc[mi][ni], false, false);
  }

  // local sum / sum^2 of pre-activations, then block reduction
  float ls = 0.0f, lq = 0.0f;
#pragma unroll
  for (int mi = 0; mi < 2; ++mi)
#pragma unroll
    for (int ni = 0; ni < 4; ++ni)
#pragma unroll
      for (int g = 0; g < 8; ++g) {
        float v = acc[mi][ni][g];
        ls += v;
        lq += v * v;
      }

  __shared__ float red[512];
  red[threadIdx.x]       = ls;
  red[256 + threadIdx.x] = lq;
  __syncthreads();
  for (int s = 128; s > 0; s >>= 1) {
    if ((int)threadIdx.x < s) {
      red[threadIdx.x]       += red[threadIdx.x + s];
      red[256 + threadIdx.x] += red[256 + threadIdx.x + s];
    }
    __syncthreads();
  }
  if (threadIdx.x == 0) {
    const int bl = z * 16 + blockIdx.y * 4 + blockIdx.x;
    psum[bl]   = red[0];
    psumsq[bl] = red[256];
  }

  // activation + store
#pragma unroll
  for (int mi = 0; mi < 2; ++mi)
#pragma unroll
    for (int ni = 0; ni < 4; ++ni)
#pragma unroll
      for (int g = 0; g < 8; ++g) {
        float v = acc[mi][ni][g];
        float a = (z == 2) ? tanhf(v) : sigm(v);
        Sact[(size_t)z * PLANE +
             (size_t)(m0 + mi * 16 + half * 8 + g) * HID + n0 + ni * 16 + r] = a;
      }
}

// ---------------------------------------------------------------------------
// Layer-0 elementwise: gates -> (c, h0), and stash activated gates (bf16)
// as A-operands for the mixing GEMM.
// ---------------------------------------------------------------------------
__global__ __launch_bounds__(256)
void ew_layer0_kernel(const float* __restrict__ G, float* __restrict__ c,
                      __bf16* __restrict__ h0, __bf16* __restrict__ g0)
{
  const int idx = blockIdx.x * 256 + threadIdx.x;
  if (idx >= PLANE) return;
  const int m = idx >> 9, n = idx & 511;
  const float* row = G + (size_t)m * G4H;
  const float i0 = sigm(row[n]);
  const float f0 = sigm(row[512 + n]);
  const float cg = tanhf(row[1024 + n]);
  const float o0 = sigm(row[1536 + n]);
  const float cc = f0 * c[idx] + i0 * cg;
  c[idx]  = cc;
  h0[idx] = (__bf16)(o0 * tanhf(cc));
  g0[idx]             = (__bf16)i0;
  g0[PLANE + idx]     = (__bf16)f0;
  g0[2 * PLANE + idx] = (__bf16)cg;
  g0[3 * PLANE + idx] = (__bf16)o0;
}

// ---------------------------------------------------------------------------
// Final elementwise per step: layer-1 state update, carry h (bf16), fused
// H->1 output projection (row reduction), and variance accumulation.
// One block per row m.
// ---------------------------------------------------------------------------
__global__ __launch_bounds__(256)
void ew_final_kernel(const float* __restrict__ Sact, float* __restrict__ c,
                     __bf16* __restrict__ hA, const float* __restrict__ W_out,
                     const float* __restrict__ b_out, float* __restrict__ out_t,
                     const float* __restrict__ psum,
                     const float* __restrict__ psumsq,
                     float* __restrict__ s_acc)
{
  const int m = blockIdx.x, tid = threadIdx.x;
  float local = 0.0f;
  for (int n = tid; n < HID; n += 256) {
    const size_t idx = (size_t)m * HID + n;
    const float i1 = Sact[idx];
    const float f1 = Sact[PLANE + idx];
    const float c1 = Sact[2 * PLANE + idx];
    const float o1 = Sact[3 * PLANE + idx];
    const float cc = f1 * c[idx] + i1 * c1;
    c[idx] = cc;
    const float hv = o1 * tanhf(cc);
    hA[idx] = (__bf16)hv;
    local += hv * W_out[n];
  }
  __shared__ float red[256];
  red[tid] = local;
  __syncthreads();
  for (int s = 128; s > 0; s >>= 1) {
    if (tid < s) red[tid] += red[tid + s];
    __syncthreads();
  }
  if (tid == 0) out_t[m] = red[0] + b_out[0];

  if (m == 0 && tid == 0) {
    const float invN = 1.0f / (float)PLANE;
    float sv = 0.0f;
    for (int z = 0; z < 4; ++z) {
      float su = 0.0f, sq = 0.0f;
      for (int b2 = 0; b2 < 16; ++b2) {
        su += psum[z * 16 + b2];
        sq += psumsq[z * 16 + b2];
      }
      const float mean = su * invN;
      sv += sq * invN - mean * mean;
    }
    *s_acc += 0.25f * sv;
  }
}

// ---------------------------------------------------------------------------
// One-time kernels
// ---------------------------------------------------------------------------
__global__ __launch_bounds__(256)
void input_proj_kernel(const float* __restrict__ x, const float* __restrict__ W_in,
                       const float* __restrict__ b_in, __bf16* __restrict__ x0)
{
  const size_t idx = (size_t)blockIdx.x * 256 + threadIdx.x;  // NT*B*H
  if (idx >= (size_t)NT * BSZ * HID) return;
  const int hcol = (int)(idx & 511);
  const size_t row = idx >> 9;
  const float* xr = x + row * NX;
  const float* wr = W_in + (size_t)hcol * NX;
  float acc = b_in[hcol];
#pragma unroll
  for (int k = 0; k < NX; ++k) acc += xr[k] * wr[k];
  x0[idx] = (__bf16)fmaxf(acc, 0.0f);
}

__global__ __launch_bounds__(256)
void pack_kernel(const float* __restrict__ W_ih0, const float* __restrict__ W_hh0,
                 const float* __restrict__ W_ih1, const float* __restrict__ W_hh1,
                 const float* __restrict__ b_ih0, const float* __restrict__ b_hh0,
                 const float* __restrict__ b_ih1, const float* __restrict__ b_hh1,
                 __bf16* __restrict__ Wc0, __bf16* __restrict__ Wc1,
                 float* __restrict__ cb0, float* __restrict__ cb1)
{
  const int idx = blockIdx.x * 256 + threadIdx.x;  // 2048*1024
  if (idx >= G4H * 1024) return;
  const int j = idx >> 10, k = idx & 1023;
  const float w0 = (k < HID) ? W_ih0[(size_t)j * HID + k]
                             : W_hh0[(size_t)j * HID + (k - HID)];
  const float w1 = (k < HID) ? W_ih1[(size_t)j * HID + k]
                             : W_hh1[(size_t)j * HID + (k - HID)];
  Wc0[idx] = (__bf16)w0;
  Wc1[idx] = (__bf16)w1;
  if (idx < G4H) {
    cb0[idx] = b_ih0[idx] + b_hh0[idx];
    cb1[idx] = b_ih1[idx] + b_hh1[idx];
  }
}

__global__ __launch_bounds__(256)
void init_state_kernel(__bf16* __restrict__ hA, float* __restrict__ c,
                       float* __restrict__ s_acc)
{
  const int idx = blockIdx.x * 256 + threadIdx.x;
  if (idx < PLANE) {
    hA[idx] = (__bf16)0.0f;
    c[idx]  = 0.0f;
  }
  if (idx == 0) *s_acc = 0.0f;
}

__global__ void finalize_kernel(const float* __restrict__ s_acc,
                                float* __restrict__ out_norm)
{
  if (threadIdx.x == 0 && blockIdx.x == 0) *out_norm = *s_acc / (float)NT;
}

// ---------------------------------------------------------------------------
// Host-side orchestration
// ---------------------------------------------------------------------------
extern "C" void kernel_launch(void* const* d_in, const int* in_sizes, int n_in,
                              void* d_out, int out_size, void* d_ws, size_t ws_size,
                              hipStream_t stream)
{
  const float* x     = (const float*)d_in[0];
  const float* W_in  = (const float*)d_in[1];
  const float* b_in  = (const float*)d_in[2];
  const float* W_ih0 = (const float*)d_in[3];
  const float* b_ih0 = (const float*)d_in[4];
  const float* W_hh0 = (const float*)d_in[5];
  const float* b_hh0 = (const float*)d_in[6];
  const float* W_ih1 = (const float*)d_in[7];
  const float* b_ih1 = (const float*)d_in[8];
  const float* W_hh1 = (const float*)d_in[9];
  const float* b_hh1 = (const float*)d_in[10];
  const float* W_out = (const float*)d_in[11];
  const float* b_out = (const float*)d_in[12];
  float* out = (float*)d_out;

  // workspace carve-up (256B aligned)
  char* ws = (char*)d_ws;
  size_t off = 0;
  auto carve = [&](size_t bytes) -> void* {
    void* p = ws + off;
    off += (bytes + 255) & ~(size_t)255;
    return p;
  };
  __bf16* x0   = (__bf16*)carve((size_t)NT * PLANE * sizeof(__bf16));  // 191 MB
  __bf16* Wc0  = (__bf16*)carve((size_t)G4H * 1024 * sizeof(__bf16));
  __bf16* Wc1  = (__bf16*)carve((size_t)G4H * 1024 * sizeof(__bf16));
  float*  cb0  = (float*)carve(G4H * sizeof(float));
  float*  cb1  = (float*)carve(G4H * sizeof(float));
  __bf16* hA   = (__bf16*)carve((size_t)PLANE * sizeof(__bf16));  // carried h
  __bf16* h0   = (__bf16*)carve((size_t)PLANE * sizeof(__bf16));  // layer-0 h
  float*  cst  = (float*)carve((size_t)PLANE * sizeof(float));    // carried c
  float*  G    = (float*)carve((size_t)BSZ * G4H * sizeof(float));
  __bf16* g0   = (__bf16*)carve((size_t)4 * PLANE * sizeof(__bf16));
  float*  Sact = (float*)carve((size_t)4 * PLANE * sizeof(float));
  float*  psum = (float*)carve(64 * sizeof(float));
  float*  psq  = (float*)carve(64 * sizeof(float));
  float*  s_acc= (float*)carve(sizeof(float));
  (void)ws_size; (void)in_sizes; (void)n_in; (void)out_size;

  // setup
  pack_kernel<<<(G4H * 1024 + 255) / 256, 256, 0, stream>>>(
      W_ih0, W_hh0, W_ih1, W_hh1, b_ih0, b_hh0, b_ih1, b_hh1, Wc0, Wc1, cb0, cb1);
  init_state_kernel<<<(PLANE + 255) / 256, 256, 0, stream>>>(hA, cst, s_acc);

  // x0[t] = relu(x_t @ W_in^T + b_in) for all t (batched, fully parallel)
  {
    const size_t n = (size_t)NT * PLANE;
    input_proj_kernel<<<(unsigned)((n + 255) / 256), 256, 0, stream>>>(
        x, W_in, b_in, x0);
  }

  const dim3 gemmGrid(G4H / 128, BSZ / 128);   // (16,4) for [512,2048]
  const dim3 sGrid(HID / 128, BSZ / 128, 4);   // (4,4,4)

  for (int t = 0; t < NT; ++t) {
    const __bf16* x0t = x0 + (size_t)t * PLANE;
    // layer-0 gates: G = [x0_t | h] @ [W_ih0|W_hh0]^T + (b_ih0+b_hh0)
    gemm_fused_bt_kernel<<<gemmGrid, 256, 0, stream>>>(
        x0t, hA, HID, HID, Wc0, 1024, cb0, G, G4H, 1024);
    ew_layer0_kernel<<<PLANE / 256, 256, 0, stream>>>(G, cst, h0, g0);
    // layer-1 raw gates: G = [x0_t | h0] @ [W_ih1|W_hh1]^T + (b_ih1+b_hh1)
    gemm_fused_bt_kernel<<<gemmGrid, 256, 0, stream>>>(
        x0t, h0, HID, HID, Wc1, 1024, cb1, G, G4H, 1024);
    // S_z = gate0_z @ gate1raw_z^T ; activations + variance partials
    sgemm_mix_kernel<<<sGrid, 256, 0, stream>>>(g0, G, Sact, psum, psq);
    // state update + fused output projection + variance accumulation
    ew_final_kernel<<<BSZ, 256, 0, stream>>>(
        Sact, cst, hA, W_out, b_out, out + (size_t)t * BSZ, psum, psq, s_acc);
  }

  finalize_kernel<<<1, 1, 0, stream>>>(s_acc, out + (size_t)NT * BSZ);
}
